// nltClassifier_18562848653807
// MI455X (gfx1250) — matrix operand order repaired and checked by reading, not compiled, not run on hardware
//
#include <hip/hip_runtime.h>

typedef _Float16 f16;
typedef __attribute__((ext_vector_type(16))) _Float16 v16h;
typedef __attribute__((ext_vector_type(8)))  _Float16 v8h;
typedef __attribute__((ext_vector_type(4)))  _Float16 v4h;
typedef __attribute__((ext_vector_type(8)))  float    v8f;

#define S_DIM 512
#define B_DIM 64
#define P_DIM 128
#define K_DIM 8
#define D_DIM 1024
#define NEG_INF (-3.402823466e38f)

// ---------------------------------------------------------------------------
// Kernel 0: build padded, N-major f16 weight matrix Wt[16][4096] from W[4096][2]
// Lane layout target (WMMA B 32x16 f16): column n's K values are contiguous.
// ---------------------------------------------------------------------------
__global__ __launch_bounds__(256) void wt_init_kernel(const float* __restrict__ W,
                                                      f16* __restrict__ wt) {
    const int i = blockIdx.x * 256 + threadIdx.x;   // 16*4096 = 65536 elements
    const int n = i >> 12;                          // 0..15
    const int k = i & 4095;                         // 0..4095
    wt[i] = (n < 2) ? (f16)W[k * 2 + n] : (f16)0.0f;
}

// ---------------------------------------------------------------------------
// Kernel 1: masked gather max-pool. One block per (b,p) row; thread t owns
// 4 consecutive d. Each gather reads a contiguous 4KB row -> fully coalesced.
// Results stored as f16 [8192][1024] for en and fr.
// ---------------------------------------------------------------------------
__global__ __launch_bounds__(256) void pool_kernel(
    const float* __restrict__ tensor,            // [S,B,D]
    const int*   __restrict__ en_idx,            // [B,P,K]
    const int*   __restrict__ fr_idx,            // [B,P,K]
    const unsigned char* __restrict__ en_mask,   // [B,P,K] bool
    const unsigned char* __restrict__ fr_mask,   // [B,P,K] bool
    const int*   __restrict__ len1,              // [B]
    f16* __restrict__ enh,                       // [B*P][D]
    f16* __restrict__ frh)                       // [B*P][D]
{
    const int r    = blockIdx.x;            // 0..8191  (= b*P + p)
    const int b    = r >> 7;                // P = 128
    const int d    = threadIdx.x << 2;      // 0..1020
    const int base = r << 3;                // K = 8

    float e0 = NEG_INF, e1 = NEG_INF, e2 = NEG_INF, e3 = NEG_INF;
#pragma unroll
    for (int k = 0; k < K_DIM; ++k) {
        if (en_mask[base + k]) {
            const size_t off = (((size_t)en_idx[base + k] * B_DIM + b) << 10) + d;
            const float4 g = *(const float4*)(tensor + off);
            e0 = fmaxf(e0, g.x); e1 = fmaxf(e1, g.y);
            e2 = fmaxf(e2, g.z); e3 = fmaxf(e3, g.w);
        }
    }

    const int loff = len1[b];
    float f0 = NEG_INF, f1 = NEG_INF, f2 = NEG_INF, f3 = NEG_INF;
#pragma unroll
    for (int k = 0; k < K_DIM; ++k) {
        if (fr_mask[base + k]) {
            const size_t off = (((size_t)(fr_idx[base + k] + loff) * B_DIM + b) << 10) + d;
            const float4 g = *(const float4*)(tensor + off);
            f0 = fmaxf(f0, g.x); f1 = fmaxf(f1, g.y);
            f2 = fmaxf(f2, g.z); f3 = fmaxf(f3, g.w);
        }
    }

    const size_t o = ((size_t)r << 10) + d;
    v4h ev = { (f16)e0, (f16)e1, (f16)e2, (f16)e3 };
    v4h fv = { (f16)f0, (f16)f1, (f16)f2, (f16)f3 };
    *(v4h*)(enh + o) = ev;
    *(v4h*)(frh + o) = fv;
}

// ---------------------------------------------------------------------------
// Kernel 2: projection GEMM via V_WMMA_F32_16X16X32_F16.
// One wave (32 lanes) per 16-row tile. K = 4096 = 4 regions x 1024:
//   region 0: en, 1: fr, 2: |en-fr|, 3: en*fr   (regenerated on the fly)
// A 16x32 f16 layout: lane l -> row l%16; halfs[0..7]=K(hi*8..+7),
// halfs[8..15]=K(hi*8+16..+23), hi = l/16.
// B 32x16 f16 layout: lane l -> col l%16; same K order as A.
// C/D f32: lane l -> N=l%16, M = v + 8*hi for VGPR v.
// ---------------------------------------------------------------------------
template <int REGION>
__device__ inline v8f region_mma(const f16* __restrict__ er,
                                 const f16* __restrict__ fr,
                                 const f16* __restrict__ wr,
                                 int hi, v8f acc)
{
    for (int kt = 0; kt < 32; ++kt) {
        const int d0 = kt * 32 + hi * 8;
        v8h a0, a1;
        if (REGION == 0) {
            a0 = *(const v8h*)(er + d0);
            a1 = *(const v8h*)(er + d0 + 16);
        } else if (REGION == 1) {
            a0 = *(const v8h*)(fr + d0);
            a1 = *(const v8h*)(fr + d0 + 16);
        } else {
            const v8h e0 = *(const v8h*)(er + d0);
            const v8h e1 = *(const v8h*)(er + d0 + 16);
            const v8h f0 = *(const v8h*)(fr + d0);
            const v8h f1 = *(const v8h*)(fr + d0 + 16);
#pragma unroll
            for (int i = 0; i < 8; ++i) {
                if (REGION == 2) {
                    f16 t0 = (f16)(e0[i] - f0[i]);
                    f16 t1 = (f16)(e1[i] - f1[i]);
                    a0[i] = (t0 < (f16)0.0f) ? (f16)(-t0) : t0;
                    a1[i] = (t1 < (f16)0.0f) ? (f16)(-t1) : t1;
                } else { // REGION == 3
                    a0[i] = (f16)(e0[i] * f0[i]);
                    a1[i] = (f16)(e1[i] * f1[i]);
                }
            }
        }
        v16h A;
#pragma unroll
        for (int i = 0; i < 8; ++i) { A[i] = a0[i]; A[i + 8] = a1[i]; }

        const v8h b0 = *(const v8h*)(wr + REGION * 1024 + kt * 32);
        const v8h b1 = *(const v8h*)(wr + REGION * 1024 + kt * 32 + 16);
        v16h Bm;
#pragma unroll
        for (int i = 0; i < 8; ++i) { Bm[i] = b0[i]; Bm[i + 8] = b1[i]; }

        acc = __builtin_amdgcn_wmma_f32_16x16x32_f16(
            /*neg_a=*/false, A, /*neg_b=*/false, Bm,
            /*c_mod=*/(short)0, acc, /*reuse_a=*/false, /*reuse_b=*/false);
    }
    return acc;
}

__global__ __launch_bounds__(32) void gemm_kernel(
    const f16* __restrict__ enh,   // [8192][1024]
    const f16* __restrict__ frh,   // [8192][1024]
    const f16* __restrict__ wt,    // [16][4096] N-major
    const float* __restrict__ bias,
    float* __restrict__ out)       // [8192][2]
{
    const int tile = blockIdx.x;          // 512 tiles of 16 rows
    const int lane = threadIdx.x & 31;
    const int m    = lane & 15;
    const int hi   = lane >> 4;

    const size_t row = (size_t)tile * 16 + m;
    const f16* er = enh + (row << 10);
    const f16* fr = frh + (row << 10);
    const f16* wr = wt + ((size_t)m << 12) + hi * 8;

    v8f acc = {};
    acc = region_mma<0>(er, fr, wr, hi, acc);
    acc = region_mma<1>(er, fr, wr, hi, acc);
    acc = region_mma<2>(er, fr, wr, hi, acc);
    acc = region_mma<3>(er, fr, wr, hi, acc);

    if (m < 2) {   // only N=0,1 are real outputs (lanes 0,1 and 16,17)
        const float bb = bias[m];
#pragma unroll
        for (int v = 0; v < 8; ++v) {
            const int M = v + hi * 8;
            out[((size_t)tile * 16 + M) * 2 + m] = acc[v] + bb;
        }
    }
}

// ---------------------------------------------------------------------------
extern "C" void kernel_launch(void* const* d_in, const int* in_sizes, int n_in,
                              void* d_out, int out_size, void* d_ws, size_t ws_size,
                              hipStream_t stream) {
    (void)in_sizes; (void)n_in; (void)out_size; (void)ws_size;

    const float* tensor = (const float*)d_in[0];
    const int*   en_idx = (const int*)d_in[1];
    const int*   fr_idx = (const int*)d_in[2];
    const unsigned char* en_mask = (const unsigned char*)d_in[3];
    const unsigned char* fr_mask = (const unsigned char*)d_in[4];
    const int*   len1   = (const int*)d_in[5];
    const float* W      = (const float*)d_in[6];
    const float* bias   = (const float*)d_in[7];
    float* out = (float*)d_out;

    // workspace: enh (16 MiB) | frh (16 MiB) | Wt (128 KiB)
    f16* enh = (f16*)d_ws;
    f16* frh = enh + (size_t)8192 * 1024;
    f16* wt  = frh + (size_t)8192 * 1024;

    hipLaunchKernelGGL(wt_init_kernel, dim3(256), dim3(256), 0, stream, W, wt);
    hipLaunchKernelGGL(pool_kernel, dim3(8192), dim3(256), 0, stream,
                       tensor, en_idx, fr_idx, en_mask, fr_mask, len1, enh, frh);
    hipLaunchKernelGGL(gemm_kernel, dim3(512), dim3(32), 0, stream,
                       enh, frh, wt, bias, out);
}
